// MTMD_Domain_Task_Expert_Gate_Shared_Skip_6227702579387
// MI455X (gfx1250) — matrix operand order repaired and checked by reading, not compiled, not run on hardware
//
#include <hip/hip_runtime.h>

typedef __attribute__((ext_vector_type(8)))  _Float16 v8h;
typedef __attribute__((ext_vector_type(16))) _Float16 v16h;
typedef __attribute__((ext_vector_type(8)))  float    v8f;

// ---------------- model dims ----------------
static constexpr int FEATS = 16, VOCAB = 1000, EDIM = 16;
static constexpr int DIN = 256;   // input_emb width
static constexpr int SMID = 512;  // STAR hidden
static constexpr int SOUT = 256;  // STAR out / emb width
static constexpr int H1D = 128, H2D = 64;
static constexpr int NDOM = 3, NTSK = 2, NEXP = 8, KDT = 6;

// ---------------- per-wave LDS layout (bytes) ----------------
// fused accumulator [6][16][64] f32 (24KB) is overlaid on the two regions that
// are dead after the star layer: SKIP (8KB -> fused k=0,1) and MID (16KB -> k=2..5)
static constexpr int OFF_IN   = 0;       // [16][256] f16 input_emb, later emb
static constexpr int OFF_SKIP = 8192;    // [16][256] f16 skip      -> fused k=0..1
static constexpr int OFF_MID  = 16384;   // [16][512] f16 emb0      -> fused k=2..5
static constexpr int OFF_H1   = 32768;   // [16][128] f16 expert h1 -> towerA [16][64] f16
static constexpr int OFF_GATE = 36864;   // [6][16][8] f32 gates
static constexpr int OFF_PROB = 39936;   // [6][16] f32 probs
static constexpr int OFF_DOM  = 40320;   // [16] i32 domain ids
static constexpr int PER_WAVE = 40384;
static constexpr int WAVES    = 8;       // 128 samples / block, 256 threads, 2 waves/SIMD
static constexpr int SMEM_BYTES = WAVES * PER_WAVE;   // 323072 <= 327680 (320KB)

// ---------------- f16 weight workspace offsets (in halves) ----------------
static constexpr size_t O_SKIP = 0;                      // [256][256]
static constexpr size_t O_SLOT = O_SKIP + 256*256;       // [3][512][256] (slot_W * shared_W, transposed)
static constexpr size_t O_STAR = O_SLOT + 3*512*256;     // [256][512]
static constexpr size_t O_GATE = O_STAR + 256*512;       // [6][16][256] (N padded 8->16 with zeros)
static constexpr size_t O_EW1  = O_GATE + 6*16*256;      // [8][128][256]
static constexpr size_t O_EW2  = O_EW1  + 8*128*256;     // [8][64][128]
static constexpr size_t O_DW1  = O_EW2  + 8*64*128;      // [3][128][256]
static constexpr size_t O_DW2  = O_DW1  + 3*128*256;     // [3][64][128]
static constexpr size_t O_TW1  = O_DW2  + 3*64*128;      // [2][128][256]
static constexpr size_t O_TW2  = O_TW1  + 2*128*256;     // [2][64][128]
static constexpr size_t O_DSK  = O_TW2  + 2*64*128;      // [6][64][256]
static constexpr size_t O_TOW  = O_DSK  + 6*64*256;      // [6][64][64]

struct Params {
  const int* x;                  // [16][B]
  const float* emb;              // [16][1000][16]
  const _Float16* skipWt; const float* skip_b; const float* skip_g; const float* skip_be;
  const _Float16* slotWt; const float* slot_b; const float* shared_b;
  const _Float16* starWt; const float* star_b; const float* star_g; const float* star_be;
  const _Float16* gateWt; const float* gate_b;
  const _Float16* eW1; const float* e_b1; const float* e_g1; const float* e_be1;
  const _Float16* eW2; const float* e_b2; const float* e_g2; const float* e_be2;
  const _Float16* dW1; const float* d_b1; const float* d_g1; const float* d_be1;
  const _Float16* dW2; const float* d_b2; const float* d_g2; const float* d_be2;
  const _Float16* tW1; const float* t_b1; const float* t_g1; const float* t_be1;
  const _Float16* tW2; const float* t_b2; const float* t_g2; const float* t_be2;
  const _Float16* dskWt; const float* dsk_b; const float* dsk_g; const float* dsk_be;
  const _Float16* towWt; const float* tow_b1; const float* tow_g; const float* tow_be;
  const float* tow_W2; const float* tow_b2;
  float* out;
  int B;
};

// ---- WMMA GEMM: A [16 x 32*KT] f16 in LDS (row-major, stride lda),
// ---- Wt [16*NT x 32*KT] f16 in global, stored [N][K] so B-frags are contiguous.
// ---- B fragments flow through a PF-deep register queue; sched_group_barrier
// ---- hints force the loads of iteration i+PF to issue under the WMMA of
// ---- iteration i (1 WMMA : 2 VMEM-read interleave after a PF*2-load prologue).
template<int NT, int KT>
__device__ __forceinline__ void wgemm(const _Float16* __restrict__ A, int lda,
                                      const _Float16* __restrict__ Wt,
                                      v8f (&c)[NT], int lane) {
  v8f zero = {};
  #pragma unroll
  for (int t = 0; t < NT; ++t) c[t] = zero;
  const int arow = lane & 15;
  const int akh  = (lane >> 4) << 4;   // 0 or 16
  const int bn   = lane & 15;
  const int bkh  = (lane >> 4) << 4;   // 0 or 16

  constexpr int TOT = NT * KT;
  constexpr int PF  = (TOT >= 4) ? 4 : TOT;

  auto loadB = [&](int i) -> v16h {
    const int ks = i / NT, t = i - ks*NT;
    return *(const v16h*)(Wt + (size_t)(t*16 + bn)*lda + ks*32 + bkh);
  };
  auto loadA = [&](int ks) -> v16h {
    const _Float16* pa = A + arow*lda + ks*32 + akh;
    v8h lo = *(const v8h*)pa;
    v8h hh = *(const v8h*)(pa + 8);
    return __builtin_shufflevector(lo, hh, 0,1,2,3,4,5,6,7,8,9,10,11,12,13,14,15);
  };

  v16h bq[PF];
  #pragma unroll
  for (int i = 0; i < PF; ++i) bq[i] = loadB(i);
  v16h a = loadA(0);
  v16h a_next = a;

  #pragma unroll
  for (int i = 0; i < TOT; ++i) {
    const int ks = i / NT, t = i - ks*NT;
    if (t == 0 && ks + 1 < KT) a_next = loadA(ks + 1);   // A one k-step ahead
    v16h b = bq[i % PF];
    if (i + PF < TOT) bq[i % PF] = loadB(i + PF);        // keep PF loads in flight
    c[t] = __builtin_amdgcn_wmma_f32_16x16x32_f16(false, a, false, b, (short)0, c[t], false, false);
    if (t == NT - 1 && ks + 1 < KT) a = a_next;
  }

#if __has_builtin(__builtin_amdgcn_sched_group_barrier)
  // 0x020 = VMEM read group, 0x008 = MFMA/WMMA group
  __builtin_amdgcn_sched_group_barrier(0x020, PF*2, 0);     // prologue loads
  #pragma unroll
  for (int i = 0; i < TOT; ++i) {
    __builtin_amdgcn_sched_group_barrier(0x008, 1, 0);      // WMMA i
    if (i + PF < TOT)
      __builtin_amdgcn_sched_group_barrier(0x020, 2, 0);    // loads for i+PF
  }
#endif
}

// bias add + LayerNorm (over 16*NT features) + ReLU, in C-fragment registers.
template<int NT>
__device__ __forceinline__ void ln_relu(v8f (&c)[NT], const float* __restrict__ bptr,
                                        const float* __restrict__ gptr,
                                        const float* __restrict__ beptr, int lane) {
  const int col = lane & 15;
  float bias[NT], gg[NT], bb[NT];
  #pragma unroll
  for (int t = 0; t < NT; ++t) {
    bias[t] = bptr[t*16 + col];
    gg[t]   = gptr[t*16 + col];
    bb[t]   = beptr[t*16 + col];
  }
  float s[8], s2[8];
  #pragma unroll
  for (int v = 0; v < 8; ++v) { s[v] = 0.f; s2[v] = 0.f; }
  #pragma unroll
  for (int t = 0; t < NT; ++t)
    #pragma unroll
    for (int v = 0; v < 8; ++v) {
      float x = c[t][v] + bias[t];
      c[t][v] = x;
      s[v] += x; s2[v] += x*x;
    }
  #pragma unroll
  for (int m = 1; m <= 8; m <<= 1)
    #pragma unroll
    for (int v = 0; v < 8; ++v) {
      s[v]  += __shfl_xor(s[v],  m, 16);
      s2[v] += __shfl_xor(s2[v], m, 16);
    }
  const float invN = 1.0f / (float)(NT*16);
  #pragma unroll
  for (int v = 0; v < 8; ++v) {
    float mu = s[v]*invN;
    float var = s2[v]*invN - mu*mu;
    float rs = rsqrtf(var + 1e-5f);
    #pragma unroll
    for (int t = 0; t < NT; ++t) {
      float y = (c[t][v]-mu)*rs*gg[t] + bb[t];
      c[t][v] = fmaxf(y, 0.f);
    }
  }
}

template<int NT>
__device__ __forceinline__ void store_act(_Float16* __restrict__ dst, int ld,
                                          const v8f (&c)[NT], int lane) {
  const int col = lane & 15, hi = lane >> 4;
  #pragma unroll
  for (int t = 0; t < NT; ++t)
    #pragma unroll
    for (int v = 0; v < 8; ++v)
      dst[(v + hi*8)*ld + t*16 + col] = (_Float16)c[t][v];
}

// one 2-layer expert (256 -> 128 -> 64), result left in h2 fragment registers
__device__ __forceinline__ void expert2(const _Float16* emb, _Float16* h1buf,
    const _Float16* W1, const float* b1, const float* g1, const float* be1,
    const _Float16* W2, const float* b2, const float* g2, const float* be2,
    v8f (&h2)[4], int lane) {
  v8f h1c[8];
  wgemm<8,8>(emb, DIN, W1, h1c, lane);
  ln_relu<8>(h1c, b1, g1, be1, lane);
  store_act<8>(h1buf, H1D, h1c, lane);
  wgemm<4,4>(h1buf, H1D, W2, h2, lane);
  ln_relu<4>(h2, b2, g2, be2, lane);
}

__global__ void __launch_bounds__(WAVES*32, 1) mtmd_fused(Params p) {
  extern __shared__ unsigned char smem[];
  const int lane = threadIdx.x & 31;
  const int wave = threadIdx.x >> 5;
  const int col  = lane & 15;
  const int hi   = lane >> 4;
  unsigned char* my = smem + wave * PER_WAVE;
  _Float16* bufIn   = (_Float16*)(my + OFF_IN);    // input_emb, later emb
  _Float16* bufSkip = (_Float16*)(my + OFF_SKIP);
  _Float16* bufMid  = (_Float16*)(my + OFF_MID);
  _Float16* h1      = (_Float16*)(my + OFF_H1);
  float*    gates   = (float*)   (my + OFF_GATE);
  float*    probs   = (float*)   (my + OFF_PROB);
  int*      domLds  = (int*)     (my + OFF_DOM);
  // fused [6][16][64] f32 overlaid on SKIP (k=0,1) and MID (k=2..5)
  auto fusedPtr = [&](int k) -> float* {
    return (k < 2) ? (float*)(my + OFF_SKIP) + (size_t)k*16*H2D
                   : (float*)(my + OFF_MID)  + (size_t)(k-2)*16*H2D;
  };

  const int s0 = (blockIdx.x * WAVES + wave) * 16;

  // CDNA5 prefetch of the weight streams (global_prefetch_b8)
  if (lane == 0) {
    __builtin_prefetch(p.skipWt, 0, 0);
    __builtin_prefetch(p.slotWt, 0, 0);
    __builtin_prefetch(p.starWt, 0, 0);
    __builtin_prefetch(p.eW1,    0, 0);
  }

  // ---- 1. embedding gather -> input_emb f16 tile [16][256] ----
  {
    const int samp = s0 + col;
    #pragma unroll 4
    for (int f = 0; f < FEATS; ++f) {
      const int idx = p.x[f * p.B + samp];
      const float* e = p.emb + ((size_t)(f*VOCAB + idx))*EDIM + hi*8;
      float4 a = *(const float4*)e;
      float4 b = *(const float4*)(e + 4);
      _Float16* d = bufIn + col*DIN + f*EDIM + hi*8;
      d[0]=(_Float16)a.x; d[1]=(_Float16)a.y; d[2]=(_Float16)a.z; d[3]=(_Float16)a.w;
      d[4]=(_Float16)b.x; d[5]=(_Float16)b.y; d[6]=(_Float16)b.z; d[7]=(_Float16)b.w;
    }
    if (hi == 0) domLds[col] = p.x[(FEATS-1)*p.B + samp];
  }
  // keep all 8 waves of the WGP in lockstep so their identical weight streams
  // hit the shared WGP$ (cuts L2 weight traffic ~8x)
  __syncthreads();

  v8f c16[16];

  // ---- 2. skip = relu(ln(input_emb @ skip_W + b)) ----
  wgemm<16,8>(bufIn, DIN, p.skipWt, c16, lane);
  ln_relu<16>(c16, p.skip_b, p.skip_g, p.skip_be, lane);
  store_act<16>(bufSkip, SOUT, c16, lane);
  __syncthreads();

  // ---- 3. STAR slot layer: per-domain GEMM + masked select -> emb0 [16][512] ----
  // N=512 computed as 2 chunks of 16 tiles to keep peak VGPR pressure ~200.
  {
    int dom_v[8];
    #pragma unroll
    for (int v = 0; v < 8; ++v) dom_v[v] = domLds[v + hi*8];
    for (int d = 0; d < NDOM; ++d) {
      for (int ch = 0; ch < 2; ++ch) {
        wgemm<16,8>(bufIn, DIN,
                    p.slotWt + (size_t)d*SMID*DIN + (size_t)ch*256*DIN, c16, lane);
        #pragma unroll
        for (int t = 0; t < 16; ++t) {
          const int n = ch*256 + t*16 + col;
          const float sb = p.slot_b[d*SMID + n] + p.shared_b[n];
          #pragma unroll
          for (int v = 0; v < 8; ++v)
            if (dom_v[v] == d)
              bufMid[(v + hi*8)*SMID + n] = (_Float16)(c16[t][v] + sb);
        }
      }
      __syncthreads();
    }
  }

  // ---- 4. star: emb = relu(ln(emb0 @ star_W + b)) + skip ----
  wgemm<16,16>(bufMid, SMID, p.starWt, c16, lane);
  ln_relu<16>(c16, p.star_b, p.star_g, p.star_be, lane);
  #pragma unroll
  for (int t = 0; t < 16; ++t)
    #pragma unroll
    for (int v = 0; v < 8; ++v) {
      const int r = v + hi*8, n = t*16 + col;
      float y = c16[t][v] + (float)bufSkip[r*SOUT + n];
      bufIn[r*SOUT + n] = (_Float16)y;        // emb overwrites input_emb
    }
  const _Float16* emb = bufIn;

  // ---- zero the fused accumulator (reuses SKIP + MID regions) ----
  for (int k = 0; k < KDT; ++k)
    for (int i = lane; i < 16*H2D; i += 32) fusedPtr(k)[i] = 0.f;
  __syncthreads();

  // ---- 5. gates: softmax(emb @ gate_W[k] + b) over 8 experts ----
  for (int k = 0; k < KDT; ++k) {
    v8f cg[1];
    wgemm<1,8>(emb, DIN, p.gateWt + (size_t)k*16*DIN, cg, lane);
    const float bsf = (col < 8) ? p.gate_b[k*8 + col] : 0.f;
    float l[8], mx[8], ex[8], sm[8];
    #pragma unroll
    for (int v = 0; v < 8; ++v) { l[v] = cg[0][v] + bsf; mx[v] = l[v]; }
    #pragma unroll
    for (int m = 1; m <= 4; m <<= 1)
      #pragma unroll
      for (int v = 0; v < 8; ++v) mx[v] = fmaxf(mx[v], __shfl_xor(mx[v], m, 16));
    #pragma unroll
    for (int v = 0; v < 8; ++v) { ex[v] = __expf(l[v] - mx[v]); sm[v] = ex[v]; }
    #pragma unroll
    for (int m = 1; m <= 4; m <<= 1)
      #pragma unroll
      for (int v = 0; v < 8; ++v) sm[v] += __shfl_xor(sm[v], m, 16);
    if (col < 8)
      #pragma unroll
      for (int v = 0; v < 8; ++v)
        gates[(k*16 + v + hi*8)*NEXP + col] = ex[v]/sm[v];
  }
  __syncthreads();

  // ---- 6. MMoE experts, gated accumulation into fused[6][16][64] ----
  for (int e = 0; e < NEXP; ++e) {
    v8f h2[4];
    expert2(emb, h1,
            p.eW1 + (size_t)e*H1D*DIN, p.e_b1 + e*H1D, p.e_g1 + e*H1D, p.e_be1 + e*H1D,
            p.eW2 + (size_t)e*H2D*H1D, p.e_b2 + e*H2D, p.e_g2 + e*H2D, p.e_be2 + e*H2D,
            h2, lane);
    for (int k = 0; k < KDT; ++k) {
      float* fk = fusedPtr(k);
      float gv[8];
      #pragma unroll
      for (int v = 0; v < 8; ++v) gv[v] = gates[(k*16 + v + hi*8)*NEXP + e];
      #pragma unroll
      for (int t = 0; t < 4; ++t)
        #pragma unroll
        for (int v = 0; v < 8; ++v)
          fk[(v + hi*8)*H2D + t*16 + col] += gv[v]*h2[t][v];
    }
    __syncthreads();
  }
  // ---- 7. domain experts ----
  for (int d = 0; d < NDOM; ++d) {
    v8f h2[4];
    expert2(emb, h1,
            p.dW1 + (size_t)d*H1D*DIN, p.d_b1 + d*H1D, p.d_g1 + d*H1D, p.d_be1 + d*H1D,
            p.dW2 + (size_t)d*H2D*H1D, p.d_b2 + d*H2D, p.d_g2 + d*H2D, p.d_be2 + d*H2D,
            h2, lane);
    for (int k = d; k < KDT; k += NDOM) {
      float* fk = fusedPtr(k);
      #pragma unroll
      for (int t = 0; t < 4; ++t)
        #pragma unroll
        for (int v = 0; v < 8; ++v)
          fk[(v + hi*8)*H2D + t*16 + col] += h2[t][v];
    }
    __syncthreads();
  }
  // ---- 8. task experts ----
  for (int ts = 0; ts < NTSK; ++ts) {
    v8f h2[4];
    expert2(emb, h1,
            p.tW1 + (size_t)ts*H1D*DIN, p.t_b1 + ts*H1D, p.t_g1 + ts*H1D, p.t_be1 + ts*H1D,
            p.tW2 + (size_t)ts*H2D*H1D, p.t_b2 + ts*H2D, p.t_g2 + ts*H2D, p.t_be2 + ts*H2D,
            h2, lane);
    for (int k = ts; k < KDT; k += NTSK) {
      float* fk = fusedPtr(k);
      #pragma unroll
      for (int t = 0; t < 4; ++t)
        #pragma unroll
        for (int v = 0; v < 8; ++v)
          fk[(v + hi*8)*H2D + t*16 + col] += h2[t][v];
    }
    __syncthreads();
  }
  // ---- 9. dskip heads: 0.5 * relu(ln(emb @ dskip_W[k])) ----
  for (int k = 0; k < KDT; ++k) {
    v8f ds[4];
    wgemm<4,8>(emb, DIN, p.dskWt + (size_t)k*H2D*DIN, ds, lane);
    ln_relu<4>(ds, p.dsk_b + k*H2D, p.dsk_g + k*H2D, p.dsk_be + k*H2D, lane);
    float* fk = fusedPtr(k);
    #pragma unroll
    for (int t = 0; t < 4; ++t)
      #pragma unroll
      for (int v = 0; v < 8; ++v)
        fk[(v + hi*8)*H2D + t*16 + col] += 0.5f*ds[t][v];
    __syncthreads();
  }

  // ---- 10. towers ----
  _Float16* towerA = h1;   // reuse
  for (int k = 0; k < KDT; ++k) {
    const float* fk = fusedPtr(k);
    #pragma unroll
    for (int t = 0; t < 4; ++t)
      #pragma unroll
      for (int v = 0; v < 8; ++v)
        towerA[(v + hi*8)*H2D + t*16 + col] =
            (_Float16)fk[(v + hi*8)*H2D + t*16 + col];
    v8f hc[4];
    wgemm<4,2>(towerA, H2D, p.towWt + (size_t)k*H2D*H2D, hc, lane);
    ln_relu<4>(hc, p.tow_b1 + k*H2D, p.tow_g + k*H2D, p.tow_be + k*H2D, lane);
    float w2[4];
    #pragma unroll
    for (int t = 0; t < 4; ++t) w2[t] = p.tow_W2[k*H2D + t*16 + col];
    float pd[8];
    #pragma unroll
    for (int v = 0; v < 8; ++v) {
      pd[v] = 0.f;
      #pragma unroll
      for (int t = 0; t < 4; ++t) pd[v] += hc[t][v]*w2[t];
    }
    #pragma unroll
    for (int m = 1; m <= 8; m <<= 1)
      #pragma unroll
      for (int v = 0; v < 8; ++v) pd[v] += __shfl_xor(pd[v], m, 16);
    const float b2 = p.tow_b2[k];
    if (col == 0)
      #pragma unroll
      for (int v = 0; v < 8; ++v)
        probs[k*16 + v + hi*8] = 1.f/(1.f + __expf(-(pd[v] + b2)));
  }

  // ---- 11. output: out[b][t] = probs[t*3 + dom(b)] ----
  if (lane < 16) {
    const int dmn  = domLds[lane];
    const int samp = s0 + lane;
    p.out[(size_t)samp*NTSK + 0] = probs[(0*NDOM + dmn)*16 + lane];
    p.out[(size_t)samp*NTSK + 1] = probs[(1*NDOM + dmn)*16 + lane];
  }
}

// -------- prep kernels: f32 [K][N] -> f16 [Ndst][K] (zero-padded cols) --------
__global__ void k_transpose_f16(const float* __restrict__ src, _Float16* __restrict__ dst,
                                int K, int N, int Ndst) {
  int i = blockIdx.x*blockDim.x + threadIdx.x;
  if (i >= K*Ndst) return;
  int n = i / K, k = i - n*K;
  float v = (n < N) ? src[(size_t)k*N + n] : 0.f;
  dst[i] = (_Float16)v;
}

__global__ void k_fuse_slot(const float* __restrict__ slotW, const float* __restrict__ sharedW,
                            _Float16* __restrict__ dst) {
  int i = blockIdx.x*blockDim.x + threadIdx.x;
  if (i >= NDOM*SMID*DIN) return;
  int d = i / (SMID*DIN);
  int r = i - d*(SMID*DIN);
  int n = r / DIN;
  int k = r - n*DIN;
  dst[i] = (_Float16)(slotW[((size_t)d*DIN + k)*SMID + n] * sharedW[(size_t)k*SMID + n]);
}

extern "C" void kernel_launch(void* const* d_in, const int* in_sizes, int n_in,
                              void* d_out, int out_size, void* d_ws, size_t ws_size,
                              hipStream_t stream) {
  auto f32 = [&](int i) { return (const float*)d_in[i]; };
  _Float16* ws = (_Float16*)d_ws;
  const int B = in_sizes[0] / FEATS;

  auto T = [&](const float* src, _Float16* dst, int K, int N, int Ndst) {
    int total = K*Ndst;
    k_transpose_f16<<<(total + 255)/256, 256, 0, stream>>>(src, dst, K, N, Ndst);
  };

  // ---- weight prep (f32 -> transposed f16 in workspace) ----
  T(f32(2),  ws + O_SKIP, DIN, SOUT, SOUT);                                   // skip_W
  k_fuse_slot<<<(NDOM*SMID*DIN + 255)/256, 256, 0, stream>>>(f32(8), f32(6), ws + O_SLOT);
  T(f32(10), ws + O_STAR, SMID, SOUT, SOUT);                                  // star_W
  for (int k = 0; k < KDT; ++k)
    T(f32(42) + (size_t)k*DIN*NEXP, ws + O_GATE + (size_t)k*16*DIN, DIN, NEXP, 16);
  for (int e = 0; e < NEXP; ++e) {
    T(f32(18) + (size_t)e*DIN*H1D, ws + O_EW1 + (size_t)e*H1D*DIN, DIN, H1D, H1D);
    T(f32(22) + (size_t)e*H1D*H2D, ws + O_EW2 + (size_t)e*H2D*H1D, H1D, H2D, H2D);
  }
  for (int d = 0; d < NDOM; ++d) {
    T(f32(26) + (size_t)d*DIN*H1D, ws + O_DW1 + (size_t)d*H1D*DIN, DIN, H1D, H1D);
    T(f32(30) + (size_t)d*H1D*H2D, ws + O_DW2 + (size_t)d*H2D*H1D, H1D, H2D, H2D);
  }
  for (int t = 0; t < NTSK; ++t) {
    T(f32(34) + (size_t)t*DIN*H1D, ws + O_TW1 + (size_t)t*H1D*DIN, DIN, H1D, H1D);
    T(f32(38) + (size_t)t*H1D*H2D, ws + O_TW2 + (size_t)t*H2D*H1D, H1D, H2D, H2D);
  }
  for (int k = 0; k < KDT; ++k) {
    T(f32(14) + (size_t)k*DIN*H2D, ws + O_DSK + (size_t)k*H2D*DIN, DIN, H2D, H2D);
    T(f32(44) + (size_t)k*H2D*H2D, ws + O_TOW + (size_t)k*H2D*H2D, H2D, H2D, H2D);
  }

  // ---- fused forward ----
  Params p;
  p.x = (const int*)d_in[0];
  p.emb = f32(1);
  p.skipWt = ws + O_SKIP; p.skip_b = f32(3); p.skip_g = f32(4); p.skip_be = f32(5);
  p.slotWt = ws + O_SLOT; p.slot_b = f32(9); p.shared_b = f32(7);
  p.starWt = ws + O_STAR; p.star_b = f32(11); p.star_g = f32(12); p.star_be = f32(13);
  p.gateWt = ws + O_GATE; p.gate_b = f32(43);
  p.eW1 = ws + O_EW1; p.e_b1 = f32(19); p.e_g1 = f32(20); p.e_be1 = f32(21);
  p.eW2 = ws + O_EW2; p.e_b2 = f32(23); p.e_g2 = f32(24); p.e_be2 = f32(25);
  p.dW1 = ws + O_DW1; p.d_b1 = f32(27); p.d_g1 = f32(28); p.d_be1 = f32(29);
  p.dW2 = ws + O_DW2; p.d_b2 = f32(31); p.d_g2 = f32(32); p.d_be2 = f32(33);
  p.tW1 = ws + O_TW1; p.t_b1 = f32(35); p.t_g1 = f32(36); p.t_be1 = f32(37);
  p.tW2 = ws + O_TW2; p.t_b2 = f32(39); p.t_g2 = f32(40); p.t_be2 = f32(41);
  p.dskWt = ws + O_DSK; p.dsk_b = f32(15); p.dsk_g = f32(16); p.dsk_be = f32(17);
  p.towWt = ws + O_TOW; p.tow_b1 = f32(45); p.tow_g = f32(46); p.tow_be = f32(47);
  p.tow_W2 = f32(48); p.tow_b2 = f32(49);
  p.out = (float*)d_out;
  p.B = B;

  hipFuncSetAttribute(reinterpret_cast<const void*>(mtmd_fused),
                      hipFuncAttributeMaxDynamicSharedMemorySize, SMEM_BYTES);
  const int blocks = B / (WAVES*16);
  mtmd_fused<<<blocks, WAVES*32, SMEM_BYTES, stream>>>(p);
}
